// CONCH_Prototype_Model_24240795419150
// MI455X (gfx1250) — matrix operand order repaired and checked by reading, not compiled, not run on hardware
//
#include <hip/hip_runtime.h>
#include <hip/hip_bf16.h>

typedef __bf16 bf16_t;
typedef __bf16 v16bf __attribute__((ext_vector_type(16)));
typedef __bf16 v8bf  __attribute__((ext_vector_type(8)));
typedef float  v8f   __attribute__((ext_vector_type(8)));
typedef int    v4i   __attribute__((vector_size(4 * sizeof(int))));

static constexpr int B_  = 2;
static constexpr int N_  = 32768;
static constexpr int DIN = 768;
static constexpr int D_  = 512;
static constexpr int P_  = 16;
static constexpr int C_  = 2;
static constexpr int KT  = DIN / 32;          // 24 k-chunks of 32
static constexpr int ROWTILES = (B_ * N_) / 16; // 4096 row tiles per tensor

// ---- workspace layout (bytes) ----
static constexpr size_t OFF_WT    = 0;                 // 512*768 bf16   = 786432
static constexpr size_t OFF_PN    = 786432;            // 16*512 bf16    = 16384
static constexpr size_t OFF_STATS = 802816;            // 64*2 f32       = 512
static constexpr size_t OFF_ATT   = 803840;            // 4*16*512 f32   = 131072
static constexpr size_t OFF_SC    = (size_t)1  << 20;  // 4*16*32768 f32 = 8 MB
static constexpr size_t OFF_FT    = (size_t)32 << 20;  // 4*512*32768 bf16 = 128 MB

// ---- async global->LDS helpers (CDNA5) ----
__device__ __forceinline__ void async_copy16(const void* g, void* l) {
#if __has_builtin(__builtin_amdgcn_global_load_async_to_lds_b128)
  __builtin_amdgcn_global_load_async_to_lds_b128(
      (__attribute__((address_space(1))) v4i*)(g),
      (__attribute__((address_space(3))) v4i*)(l), 0, 0);
#else
  *(float4*)l = *(const float4*)g;
#endif
}

__device__ __forceinline__ void async_wait0() {
#if __has_builtin(__builtin_amdgcn_s_wait_asynccnt)
  __builtin_amdgcn_s_wait_asynccnt(0);
#else
#if __has_builtin(__builtin_amdgcn_global_load_async_to_lds_b128)
  asm volatile("s_wait_asynccnt 0" ::: "memory");
#endif
#endif
}

// ---------------------------------------------------------------------------
// Kernel 1: W [768,512] f32 -> Wt [512,768] bf16 ; prototypes -> l2norm bf16
// ---------------------------------------------------------------------------
__global__ __launch_bounds__(256) void prep_kernel(
    const float* __restrict__ W, const float* __restrict__ proto,
    bf16_t* __restrict__ Wt, bf16_t* __restrict__ pn)
{
  __shared__ float red[8];
  __shared__ float totS;
  const int bid = blockIdx.x;
  if (bid < 1536) {
    int t = bid * 256 + threadIdx.x;        // 0 .. 393215
    int k = t >> 9, d = t & 511;
    Wt[d * DIN + k] = (bf16_t)W[t];
  } else {
    int p = bid - 1536;                     // 0..15
    const float* row = proto + p * D_;
    float s = 0.f;
    for (int i = threadIdx.x; i < D_; i += 256) { float v = row[i]; s += v * v; }
    for (int m = 16; m >= 1; m >>= 1) s += __shfl_xor(s, m, 32);
    if ((threadIdx.x & 31) == 0) red[threadIdx.x >> 5] = s;
    __syncthreads();
    if (threadIdx.x == 0) {
      float t = 0.f;
      for (int i = 0; i < 8; ++i) t += red[i];
      totS = t;
    }
    __syncthreads();
    float rinv = 1.f / fmaxf(sqrtf(totS), 1e-12f);
    for (int i = threadIdx.x; i < D_; i += 256) pn[p * D_ + i] = (bf16_t)(row[i] * rinv);
  }
}

// ---------------------------------------------------------------------------
// Kernel 2: fused projection + bias + L2 norm + featsT store + prototype scores
// grid: (4096 row-tiles, 2 tensors), block 256 (8 waves; each wave 4 col tiles)
// B operands double-buffered (ping-pong, no register rotation copies).
// ---------------------------------------------------------------------------
__global__ __launch_bounds__(256) void proj_kernel(
    const float* __restrict__ x_s, const float* __restrict__ x_l,
    const float* __restrict__ bias,
    const bf16_t* __restrict__ Wt, const bf16_t* __restrict__ pn,
    bf16_t* __restrict__ featsT, float* __restrict__ scores)
{
  __shared__ __align__(32) char smem[16 * DIN * 4];  // 48KB: f32 A tile, reused as bf16 feats
  __shared__ float rowsq[16];
  __shared__ float scoreTile[256];
  float*  Ax = (float*)smem;
  bf16_t* fl = (bf16_t*)smem;

  const int tid  = threadIdx.x;
  const int lane = tid & 31, wid = tid >> 5;
  const int row  = lane & 15, hi = lane >> 4;     // A/B lane decomposition
  const int tile = blockIdx.x, tensor = blockIdx.y;
  const int b    = tile >> 11;                    // 2048 tiles per batch
  const int nloc0 = (tile << 4) & (N_ - 1);
  const int tb   = tensor * 2 + b;

  const float* xg = (tensor ? x_l : x_s) + (size_t)tile * 16 * DIN;

  // stage 16x768 f32 tile (48KB, contiguous) into LDS with async copies
  #pragma unroll
  for (int i = 0; i < 12; ++i) {
    int ofs = (tid + i * 256) * 16;
    async_copy16((const char*)xg + ofs, smem + ofs);
  }
  if (tid < 16) rowsq[tid] = 0.f;
  scoreTile[tid] = 0.f;

  // per-wave B base: + d*DIN + k
  const bf16_t* wbase = Wt + hi * 16;
  int doffs[4];
  #pragma unroll
  for (int j = 0; j < 4; ++j) doffs[j] = ((wid * 4 + j) * 16 + row) * DIN;

  // preload B tiles for kc = 0 (overlaps with async A staging)
  v16bf b0[4], b1[4];
  #pragma unroll
  for (int j = 0; j < 4; ++j) b0[j] = *(const v16bf*)(wbase + doffs[j]);

  async_wait0();
  __syncthreads();

  v8f acc[4] = {};
  for (int kc = 0; kc < KT; kc += 2) {
    // ---- even step: consume b0, prefetch b1 for kc+1 ----
    {
      const int kb = kc * 32 + hi * 16;
      v8f x0 = *(const v8f*)&Ax[row * DIN + kb];
      v8f x1 = *(const v8f*)&Ax[row * DIN + kb + 8];
      v16bf a;
      #pragma unroll
      for (int j = 0; j < 8; ++j) { a[j] = (bf16_t)x0[j]; a[8 + j] = (bf16_t)x1[j]; }
      const int kn = (kc + 1) * 32;
      #pragma unroll
      for (int j = 0; j < 4; ++j) b1[j] = *(const v16bf*)(wbase + doffs[j] + kn);
      #pragma unroll
      for (int j = 0; j < 4; ++j)
        acc[j] = __builtin_amdgcn_wmma_f32_16x16x32_bf16(
            false, a, false, b0[j], (short)0, acc[j], false, false);
    }
    // ---- odd step: consume b1, prefetch b0 for kc+2 (wraps on last; harmless) ----
    {
      const int kb = (kc + 1) * 32 + hi * 16;
      v8f x0 = *(const v8f*)&Ax[row * DIN + kb];
      v8f x1 = *(const v8f*)&Ax[row * DIN + kb + 8];
      v16bf a;
      #pragma unroll
      for (int j = 0; j < 8; ++j) { a[j] = (bf16_t)x0[j]; a[8 + j] = (bf16_t)x1[j]; }
      const int kn = (kc + 2 < KT) ? (kc + 2) * 32 : 0;
      #pragma unroll
      for (int j = 0; j < 4; ++j) b0[j] = *(const v16bf*)(wbase + doffs[j] + kn);
      #pragma unroll
      for (int j = 0; j < 4; ++j)
        acc[j] = __builtin_amdgcn_wmma_f32_16x16x32_bf16(
            false, a, false, b1[j], (short)0, acc[j], false, false);
    }
  }

  // bias add + per-row square sums
  float rs[8];
  #pragma unroll
  for (int r = 0; r < 8; ++r) rs[r] = 0.f;
  #pragma unroll
  for (int j = 0; j < 4; ++j) {
    const int d = (wid * 4 + j) * 16 + row;
    const float bb = bias[d];
    #pragma unroll
    for (int r = 0; r < 8; ++r) { acc[j][r] += bb; rs[r] += acc[j][r] * acc[j][r]; }
  }
  #pragma unroll
  for (int r = 0; r < 8; ++r) {
    float s = rs[r];
    s += __shfl_xor(s, 1, 32); s += __shfl_xor(s, 2, 32);
    s += __shfl_xor(s, 4, 32); s += __shfl_xor(s, 8, 32);
    if (row == 0) atomicAdd(&rowsq[hi * 8 + r], s);
  }
  __syncthreads();

  float rinv[8];
  #pragma unroll
  for (int r = 0; r < 8; ++r)
    rinv[r] = 1.f / fmaxf(sqrtf(rowsq[hi * 8 + r]), 1e-12f);

  // write normalized features: global featsT (d-major, coalesced 16B packs) + LDS tile
  #pragma unroll
  for (int j = 0; j < 4; ++j) {
    const int d = (wid * 4 + j) * 16 + row;
    v8bf pk;
    #pragma unroll
    for (int r = 0; r < 8; ++r) pk[r] = (bf16_t)(acc[j][r] * rinv[r]);
    *(v8bf*)(featsT + ((size_t)tb * D_ + d) * N_ + nloc0 + hi * 8) = pk;
    #pragma unroll
    for (int r = 0; r < 8; ++r) fl[(hi * 8 + r) * D_ + d] = pk[r];
  }
  __syncthreads();

  // prototype scores: 16(rows) x 16(P) tile, K=512 split across 8 waves (64 each)
  v8f scAcc = {};
  const int k0s = wid * 64;
  #pragma unroll
  for (int kk = 0; kk < 2; ++kk) {
    const int k0 = k0s + kk * 32;
    const int kb = k0 + hi * 16;
    v8bf f0 = *(const v8bf*)&fl[row * D_ + kb];
    v8bf f1 = *(const v8bf*)&fl[row * D_ + kb + 8];
    v16bf a;
    #pragma unroll
    for (int j = 0; j < 8; ++j) { a[j] = f0[j]; a[8 + j] = f1[j]; }
    const int koff = k0 + hi * 16;
    v16bf bv2 = *(const v16bf*)&pn[row * D_ + koff];
    scAcc = __builtin_amdgcn_wmma_f32_16x16x32_bf16(
        false, a, false, bv2, (short)0, scAcc, false, false);
  }
  #pragma unroll
  for (int r = 0; r < 8; ++r)
    atomicAdd(&scoreTile[(hi * 8 + r) * 16 + row], scAcc[r]);
  __syncthreads();

  {
    const int nr = tid >> 4, p = tid & 15;
    const float v = scoreTile[tid] * 0.04419417382415922f;  // 1/sqrt(512)
    scores[((size_t)tb * P_ + p) * N_ + nloc0 + nr] = v;
  }
}

// ---------------------------------------------------------------------------
// Kernel 3: softmax stats (max, sum-exp) per (tensor,b,p); zero attended
// ---------------------------------------------------------------------------
__global__ __launch_bounds__(256) void softstat_kernel(
    const float* __restrict__ scores, float* __restrict__ stats,
    float* __restrict__ att)
{
  __shared__ float red[8];
  __shared__ float bc;
  const int r = blockIdx.x;                 // 0..63
  const float* base = scores + (size_t)r * N_;
  const int tid = threadIdx.x;

  float m = -1e30f;
  for (int i = tid; i < N_; i += 256) m = fmaxf(m, base[i]);
  for (int s = 16; s >= 1; s >>= 1) m = fmaxf(m, __shfl_xor(m, s, 32));
  if ((tid & 31) == 0) red[tid >> 5] = m;
  __syncthreads();
  if (tid == 0) {
    float t = red[0];
    for (int i = 1; i < 8; ++i) t = fmaxf(t, red[i]);
    bc = t;
  }
  __syncthreads();
  const float M = bc;
  __syncthreads();

  float s = 0.f;
  for (int i = tid; i < N_; i += 256) s += __expf(base[i] - M);
  for (int k = 16; k >= 1; k >>= 1) s += __shfl_xor(s, k, 32);
  if ((tid & 31) == 0) red[tid >> 5] = s;
  __syncthreads();
  if (tid == 0) {
    float t = 0.f;
    for (int i = 0; i < 8; ++i) t += red[i];
    stats[r * 2] = M;
    stats[r * 2 + 1] = t;
  }
  for (int i = tid; i < D_; i += 256) att[r * D_ + i] = 0.f;
}

// ---------------------------------------------------------------------------
// Kernel 4: attended[p,d] = sum_n softmax_w[p,n] * feats[n,d] via WMMA
// grid: (64 N-slices of 512, 4 tensor-batch combos); split-K atomics into att.
// B operands double-buffered (ping-pong).
// ---------------------------------------------------------------------------
__global__ __launch_bounds__(256) void pool_kernel(
    const float* __restrict__ scores, const float* __restrict__ stats,
    const bf16_t* __restrict__ featsT, float* __restrict__ att)
{
  const int tid  = threadIdx.x;
  const int lane = tid & 31, wid = tid >> 5;
  const int row  = lane & 15, hi = lane >> 4;
  const int tb   = blockIdx.y;
  const int n0base = blockIdx.x * (N_ / 64);  // 512 n per slice

  const float M    = stats[(tb * P_ + row) * 2];
  const float rinv = 1.f / stats[(tb * P_ + row) * 2 + 1];
  const float* sp  = scores + ((size_t)tb * P_ + row) * N_ + n0base;
  const bf16_t* fb = featsT + (size_t)tb * D_ * N_ + n0base + hi * 16;

  size_t foffs[4];
  #pragma unroll
  for (int j = 0; j < 4; ++j) foffs[j] = (size_t)((wid * 4 + j) * 16 + row) * N_;

  // preload B tiles for kc = 0
  v16bf b0[4], b1[4];
  #pragma unroll
  for (int j = 0; j < 4; ++j) b0[j] = *(const v16bf*)(fb + foffs[j]);

  v8f acc[4] = {};
  for (int kc = 0; kc < 16; kc += 2) {
    // ---- even step ----
    {
      const int kb = kc * 32 + hi * 16;
      v8f s0 = *(const v8f*)&sp[kb];
      v8f s1 = *(const v8f*)&sp[kb + 8];
      v16bf a;
      #pragma unroll
      for (int j = 0; j < 8; ++j) {
        a[j]     = (bf16_t)(__expf(s0[j] - M) * rinv);
        a[8 + j] = (bf16_t)(__expf(s1[j] - M) * rinv);
      }
      const int kn = (kc + 1) * 32;
      #pragma unroll
      for (int j = 0; j < 4; ++j) b1[j] = *(const v16bf*)(fb + foffs[j] + kn);
      #pragma unroll
      for (int j = 0; j < 4; ++j)
        acc[j] = __builtin_amdgcn_wmma_f32_16x16x32_bf16(
            false, a, false, b0[j], (short)0, acc[j], false, false);
    }
    // ---- odd step ----
    {
      const int kb = (kc + 1) * 32 + hi * 16;
      v8f s0 = *(const v8f*)&sp[kb];
      v8f s1 = *(const v8f*)&sp[kb + 8];
      v16bf a;
      #pragma unroll
      for (int j = 0; j < 8; ++j) {
        a[j]     = (bf16_t)(__expf(s0[j] - M) * rinv);
        a[8 + j] = (bf16_t)(__expf(s1[j] - M) * rinv);
      }
      const int kn = (kc + 2 < 16) ? (kc + 2) * 32 : 0;
      #pragma unroll
      for (int j = 0; j < 4; ++j) b0[j] = *(const v16bf*)(fb + foffs[j] + kn);
      #pragma unroll
      for (int j = 0; j < 4; ++j)
        acc[j] = __builtin_amdgcn_wmma_f32_16x16x32_bf16(
            false, a, false, b1[j], (short)0, acc[j], false, false);
    }
  }
  #pragma unroll
  for (int j = 0; j < 4; ++j) {
    const int d = (wid * 4 + j) * 16 + row;
    #pragma unroll
    for (int r = 0; r < 8; ++r)
      atomicAdd(&att[(tb * P_ + hi * 8 + r) * D_ + d], acc[j][r]);
  }
}

// ---------------------------------------------------------------------------
// Kernel 5: mean over P, l2norm, logits, log-softmax loss, Y_prob, Y_hat
// ---------------------------------------------------------------------------
__global__ __launch_bounds__(256) void head_kernel(
    const float* __restrict__ att, const float* __restrict__ tlow,
    const float* __restrict__ thigh, const int* __restrict__ label,
    float* __restrict__ out, int out_size)
{
  __shared__ float mv[4 * D_];
  __shared__ float nsq[4];
  __shared__ float lg[4];
  const int tid = threadIdx.x;
  if (tid < 4) { nsq[tid] = 0.f; lg[tid] = 0.f; }
  __syncthreads();

  for (int i = tid; i < 4 * D_; i += 256) {
    const int t = i >> 9, d = i & (D_ - 1);
    float s = 0.f;
    for (int p = 0; p < P_; ++p) s += att[(t * P_ + p) * D_ + d];
    s *= (1.f / 16.f);
    mv[i] = s;
    atomicAdd(&nsq[t], s * s);
  }
  __syncthreads();

  float rn[4];
  for (int t = 0; t < 4; ++t) rn[t] = 1.f / fmaxf(sqrtf(nsq[t]), 1e-12f);

  for (int i = tid; i < B_ * C_ * D_; i += 256) {
    const int bb = i >> 10, rem = i & 1023, c = rem >> 9, d = rem & 511;
    const float v = mv[bb * D_ + d] * rn[bb] * tlow[c * D_ + d]
                  + mv[(2 + bb) * D_ + d] * rn[2 + bb] * thigh[c * D_ + d];
    atomicAdd(&lg[bb * 2 + c], v);
  }
  __syncthreads();

  if (tid == 0) {
    float loss = 0.f, yp[4], yh[2];
    for (int bb = 0; bb < 2; ++bb) {
      const float l0 = lg[bb * 2], l1 = lg[bb * 2 + 1];
      const float mx = fmaxf(l0, l1);
      const float e0 = __expf(l0 - mx), e1 = __expf(l1 - mx);
      const float Z = e0 + e1;
      yp[bb * 2] = e0 / Z; yp[bb * 2 + 1] = e1 / Z;
      yh[bb] = (l1 > l0) ? 1.f : 0.f;
      const int lab = label[bb];
      const float lp = ((lab == 0) ? (l0 - mx) : (l1 - mx)) - __logf(Z);
      loss -= 0.5f * lp;
    }
    const float res[7] = {yp[0], yp[1], yp[2], yp[3], yh[0], yh[1], loss};
    for (int i = 0; i < 7 && i < out_size; ++i) out[i] = res[i];
  }
}

// ---------------------------------------------------------------------------
extern "C" void kernel_launch(void* const* d_in, const int* in_sizes, int n_in,
                              void* d_out, int out_size, void* d_ws, size_t ws_size,
                              hipStream_t stream) {
  (void)in_sizes; (void)n_in; (void)ws_size;
  const float* x_s   = (const float*)d_in[0];
  const float* x_l   = (const float*)d_in[2];
  const float* W     = (const float*)d_in[4];
  const float* bpr   = (const float*)d_in[5];
  const float* proto = (const float*)d_in[6];
  const float* tlow  = (const float*)d_in[7];
  const float* thigh = (const float*)d_in[8];
  const int*   label = (const int*)d_in[9];

  char* ws = (char*)d_ws;
  bf16_t* Wt    = (bf16_t*)(ws + OFF_WT);
  bf16_t* pn    = (bf16_t*)(ws + OFF_PN);
  float*  stats = (float*)(ws + OFF_STATS);
  float*  att   = (float*)(ws + OFF_ATT);
  float*  sc    = (float*)(ws + OFF_SC);
  bf16_t* ftT   = (bf16_t*)(ws + OFF_FT);

  prep_kernel<<<dim3(1536 + 16), 256, 0, stream>>>(W, proto, Wt, pn);
  proj_kernel<<<dim3(ROWTILES, 2), 256, 0, stream>>>(x_s, x_l, bpr, Wt, pn, ftT, sc);
  softstat_kernel<<<dim3(64), 256, 0, stream>>>(sc, stats, att);
  pool_kernel<<<dim3(64, 4), 256, 0, stream>>>(sc, stats, ftT, att);
  head_kernel<<<dim3(1), 256, 0, stream>>>(att, tlow, thigh, label, (float*)d_out, out_size);
}
